// MHMS_CLIP_9277129359363
// MI455X (gfx1250) — matrix operand order repaired and checked by reading, not compiled, not run on hardware
//
#include <hip/hip_runtime.h>
#include <hip/hip_bf16.h>

typedef _Float16 v8h  __attribute__((ext_vector_type(8)));
typedef _Float16 v16h __attribute__((ext_vector_type(16)));
typedef float    v8f  __attribute__((ext_vector_type(8)));

#define BATCH 16
#define SEQ   1024
#define MVID  1024
#define DIM   2560
#define ADIM  256
#define REGc  0.05f
#define EPSc  1e-8f
#define NROWS (BATCH*SEQ)      // 16384 rows for each projection

#define TILE_R 64
#define TILE_C 64
#define TILE_K 32
#define LDSS   40              // LDS row stride in halves (80B: 16B-aligned, conflict-free)

// ---------------------------------------------------------------------------
// One-time weight transpose+convert: WT[n, k] = (f16) W[k, n].
// Coalesced 2B writes; strided reads are served from L2 (W is 2.6 MB).
// ---------------------------------------------------------------------------
__global__ __launch_bounds__(256) void wtrans_kernel(const float* __restrict__ W,
                                                     _Float16* __restrict__ WT)
{
  const int idx = blockIdx.x * 256 + threadIdx.x;   // over ADIM*DIM
  const int n = idx / DIM;
  const int k = idx - n * DIM;
  WT[idx] = (_Float16)W[(size_t)k * ADIM + n];
}

// ---------------------------------------------------------------------------
// P[row, n] = (X[row, :] * prob[row]) . W[:, n] + bias[n]   (f16 WMMA, f32 acc)
// A tile staged in LDS (double-buffered, 1 barrier/K-step); B fragments loaded
// directly from the pre-transposed f16 weights (L2-resident).
// ---------------------------------------------------------------------------
__global__ __launch_bounds__(256) void proj_kernel(
    const float* __restrict__ X, const float* __restrict__ prob,
    const _Float16* __restrict__ WT, const float* __restrict__ bias,
    float* __restrict__ P)
{
  __shared__ _Float16 As[2 * TILE_R * LDSS];   // double-buffered [row][k]

  const int rb   = blockIdx.x >> 2;            // 256 row blocks
  const int cb   = blockIdx.x & 3;             // 4 col blocks
  const int row0 = rb * TILE_R;
  const int col0 = cb * TILE_C;

  const int t    = threadIdx.x;
  const int lane = t & 31;
  const int wave = t >> 5;
  const int rt   = wave & 3;                   // 4 row tiles of 16
  const int cp   = wave >> 2;                  // 2 col-pairs (2x16 cols each)

  // A-tile loader assignment: 64 rows x 32 k, 8 floats per thread
  const int ar  = t >> 2;
  const int ac8 = (t & 3) * 8;
  const float pr = prob[row0 + ar];
  const float* xrow = &X[(size_t)(row0 + ar) * DIM + ac8];
  _Float16* myslot0 = &As[ar * LDSS + ac8];

  // fragment geometry (contiguous k order k = 16*(lane/16) + s, same for A and B)
  const int fr  = lane & 15;
  const int kbA = (lane < 16) ? 0 : 16;        // A: 16 contiguous K per lane
  const int koB = (lane < 16) ? 0 : 16;        // B: 16 contiguous K per lane

  const _Float16* wcol0 = &WT[(size_t)(col0 + (cp*2+0)*16 + fr) * DIM + koB];
  const _Float16* wcol1 = &WT[(size_t)(col0 + (cp*2+1)*16 + fr) * DIM + koB];
  const _Float16* afrag_base = &As[(rt*16 + fr) * LDSS + kbA];

  v8f acc0 = {}; v8f acc1 = {};

  auto stage = [&](int buf, int k0) {
    const float4* px = (const float4*)(xrow + k0);
    float4 x0 = px[0], x1 = px[1];
    v8h a;
    a[0]=(_Float16)(x0.x*pr); a[1]=(_Float16)(x0.y*pr);
    a[2]=(_Float16)(x0.z*pr); a[3]=(_Float16)(x0.w*pr);
    a[4]=(_Float16)(x1.x*pr); a[5]=(_Float16)(x1.y*pr);
    a[6]=(_Float16)(x1.z*pr); a[7]=(_Float16)(x1.w*pr);
    *(v8h*)(myslot0 + buf * (TILE_R * LDSS)) = a;
  };

  stage(0, 0);
  __syncthreads();

  const int NK = DIM / TILE_K;                 // 80
  for (int kk = 0; kk < NK; ++kk) {
    const int cur = kk & 1;
    if (kk + 1 < NK) stage(1 - cur, (kk + 1) * TILE_K);

    const _Float16* asb = afrag_base + cur * (TILE_R * LDSS);
    v8h alo = *(const v8h*)(asb);
    v8h ahi = *(const v8h*)(asb + 8);
    v16h af = __builtin_shufflevector(alo, ahi, 0,1,2,3,4,5,6,7,8,9,10,11,12,13,14,15);

    const size_t ko = (size_t)kk * TILE_K;
    v8h b0l = *(const v8h*)(wcol0 + ko);
    v8h b0h = *(const v8h*)(wcol0 + ko + 8);
    v16h bf0 = __builtin_shufflevector(b0l, b0h, 0,1,2,3,4,5,6,7,8,9,10,11,12,13,14,15);
    v8h b1l = *(const v8h*)(wcol1 + ko);
    v8h b1h = *(const v8h*)(wcol1 + ko + 8);
    v16h bf1 = __builtin_shufflevector(b1l, b1h, 0,1,2,3,4,5,6,7,8,9,10,11,12,13,14,15);

    acc0 = __builtin_amdgcn_wmma_f32_16x16x32_f16(false, af, false, bf0, (short)0, acc0, false, false);
    acc1 = __builtin_amdgcn_wmma_f32_16x16x32_f16(false, af, false, bf1, (short)0, acc1, false, false);
    __syncthreads();
  }

  // C/D layout: VGPR i -> row i (lanes 0-15) / row i+8 (lanes 16-31), col = lane&15
  const int rbase = row0 + rt*16 + ((lane < 16) ? 0 : 8);
  const int cg0 = col0 + (cp*2+0)*16 + fr;
  const int cg1 = col0 + (cp*2+1)*16 + fr;
  const float b0 = bias[cg0], b1 = bias[cg1];
  #pragma unroll
  for (int i = 0; i < 8; ++i) {
    P[(size_t)(rbase + i) * ADIM + cg0] = acc0[i] + b0;
    P[(size_t)(rbase + i) * ADIM + cg1] = acc1[i] + b1;
  }
}

// ---------------------------------------------------------------------------
// L2-normalize each 256-wide row, write f16.  One wave per row.
// ---------------------------------------------------------------------------
__global__ __launch_bounds__(256) void norm_kernel(const float* __restrict__ P,
                                                   _Float16* __restrict__ N)
{
  const int lane = threadIdx.x & 31;
  const int wave = threadIdx.x >> 5;
  const int row  = blockIdx.x * 8 + wave;
  const float4* p = (const float4*)&P[(size_t)row * ADIM + lane * 8];
  float4 x0 = p[0], x1 = p[1];
  float ss = x0.x*x0.x + x0.y*x0.y + x0.z*x0.z + x0.w*x0.w
           + x1.x*x1.x + x1.y*x1.y + x1.z*x1.z + x1.w*x1.w;
  #pragma unroll
  for (int off = 16; off > 0; off >>= 1) ss += __shfl_xor(ss, off, 32);
  const float inv = 1.0f / fmaxf(sqrtf(ss), EPSc);
  v8h o;
  o[0]=(_Float16)(x0.x*inv); o[1]=(_Float16)(x0.y*inv);
  o[2]=(_Float16)(x0.z*inv); o[3]=(_Float16)(x0.w*inv);
  o[4]=(_Float16)(x1.x*inv); o[5]=(_Float16)(x1.y*inv);
  o[6]=(_Float16)(x1.z*inv); o[7]=(_Float16)(x1.w*inv);
  *(v8h*)&N[(size_t)row * ADIM + lane * 8] = o;
}

// ---------------------------------------------------------------------------
// Kmat[b,i,m] = exp((En[b,i,:] . Vn[b,m,:] - 1) / REG)   (f16 WMMA over A=256)
// ---------------------------------------------------------------------------
__global__ __launch_bounds__(256) void sim_kernel(const _Float16* __restrict__ En,
                                                  const _Float16* __restrict__ Vn,
                                                  float* __restrict__ Kmat)
{
  const int b  = blockIdx.x >> 8;
  const int tb = blockIdx.x & 255;
  const int i0 = (tb >> 4) * 64;
  const int m0 = (tb & 15) * 64;

  const int lane = threadIdx.x & 31;
  const int wave = threadIdx.x >> 5;
  const int rt   = wave & 3;
  const int cp   = wave >> 2;

  const _Float16* Eb = En + (size_t)b * SEQ  * ADIM;
  const _Float16* Vb = Vn + (size_t)b * MVID * ADIM;

  const int fr  = lane & 15;
  const int kbA = (lane < 16) ? 0 : 16;
  const int koB = (lane < 16) ? 0 : 16;
  const int er  = i0 + rt*16 + fr;
  const int mc0 = m0 + (cp*2+0)*16 + fr;
  const int mc1 = m0 + (cp*2+1)*16 + fr;

  v8f acc0 = {}; v8f acc1 = {};
  #pragma unroll
  for (int a0 = 0; a0 < ADIM; a0 += 32) {
    v8h alo = *(const v8h*)&Eb[(size_t)er * ADIM + a0 + kbA];
    v8h ahi = *(const v8h*)&Eb[(size_t)er * ADIM + a0 + kbA + 8];
    v16h af = __builtin_shufflevector(alo, ahi, 0,1,2,3,4,5,6,7,8,9,10,11,12,13,14,15);
    v8h b0l = *(const v8h*)&Vb[(size_t)mc0 * ADIM + a0 + koB];
    v8h b0h = *(const v8h*)&Vb[(size_t)mc0 * ADIM + a0 + koB + 8];
    v16h bf0 = __builtin_shufflevector(b0l, b0h, 0,1,2,3,4,5,6,7,8,9,10,11,12,13,14,15);
    v8h b1l = *(const v8h*)&Vb[(size_t)mc1 * ADIM + a0 + koB];
    v8h b1h = *(const v8h*)&Vb[(size_t)mc1 * ADIM + a0 + koB + 8];
    v16h bf1 = __builtin_shufflevector(b1l, b1h, 0,1,2,3,4,5,6,7,8,9,10,11,12,13,14,15);
    acc0 = __builtin_amdgcn_wmma_f32_16x16x32_f16(false, af, false, bf0, (short)0, acc0, false, false);
    acc1 = __builtin_amdgcn_wmma_f32_16x16x32_f16(false, af, false, bf1, (short)0, acc1, false, false);
  }

  float* Kb = Kmat + (size_t)b * SEQ * MVID;
  const int ir = i0 + rt*16 + ((lane < 16) ? 0 : 8);
  #pragma unroll
  for (int i = 0; i < 8; ++i) {
    Kb[(size_t)(ir + i) * MVID + mc0] = __expf((acc0[i] - 1.0f) * (1.0f / REGc));
    Kb[(size_t)(ir + i) * MVID + mc1] = __expf((acc1[i] - 1.0f) * (1.0f / REGc));
  }
}

// ---------------------------------------------------------------------------
// Sinkhorn
// ---------------------------------------------------------------------------
__global__ void init_uv_kernel(float* __restrict__ u, float* __restrict__ v)
{
  const int i = blockIdx.x * 256 + threadIdx.x;   // 16384 entries each
  u[i] = 1.0f / 1024.0f;
  v[i] = 1.0f / 1024.0f;
}

// u[b,k] = mu / (sum_m K[b,k,m] v[b,m] + eps); one wave per row
__global__ __launch_bounds__(256) void sk_row_kernel(const float* __restrict__ Kmat,
                                                     const float* __restrict__ v,
                                                     float* __restrict__ u)
{
  const int lane = threadIdx.x & 31;
  const int wave = threadIdx.x >> 5;
  const int row  = blockIdx.x * 8 + wave;          // row = b*1024 + k
  const int b    = row >> 10;
  const float* kr = Kmat + (size_t)row * MVID;
  const float* vb = v + b * MVID;
  float s = 0.0f;
  #pragma unroll 4
  for (int m = lane; m < MVID; m += 32) s += kr[m] * vb[m];
  #pragma unroll
  for (int off = 16; off > 0; off >>= 1) s += __shfl_xor(s, off, 32);
  if (lane == 0) u[row] = (1.0f / 1024.0f) / (s + EPSc);
}

// v[b,m] = nu / (sum_k K[b,k,m] u[b,k] + eps); coalesced 64-column tiles
__global__ __launch_bounds__(256) void sk_col_kernel(const float* __restrict__ Kmat,
                                                     const float* __restrict__ u,
                                                     float* __restrict__ v)
{
  __shared__ float su[SEQ];
  __shared__ float red[256];
  const int b  = blockIdx.x >> 4;
  const int m0 = (blockIdx.x & 15) * 64;
  const int t  = threadIdx.x;
  for (int i = t; i < SEQ; i += 256) su[i] = u[b * SEQ + i];
  __syncthreads();
  const int c  = t & 63;
  const int kg = t >> 6;
  const float* kb = Kmat + (size_t)b * SEQ * MVID;
  float s = 0.0f;
  const int kend = kg * 256 + 256;
  #pragma unroll 4
  for (int k = kg * 256; k < kend; ++k) s += kb[(size_t)k * MVID + m0 + c] * su[k];
  red[t] = s;
  __syncthreads();
  if (t < 64) {
    float tot = red[t] + red[t + 64] + red[t + 128] + red[t + 192];
    v[b * MVID + m0 + t] = (1.0f / 1024.0f) / (tot + EPSc);
  }
}

// ---------------------------------------------------------------------------
// T = u (.) K (.) v  -> out[1..]; loss = sum(T*C)/B with C = -REG*log(K) -> out[0]
// ---------------------------------------------------------------------------
__global__ void zero_loss_kernel(float* __restrict__ out) { out[0] = 0.0f; }

__global__ __launch_bounds__(256) void finalize_kernel(const float* __restrict__ Kmat,
                                                       const float* __restrict__ u,
                                                       const float* __restrict__ v,
                                                       float* __restrict__ out)
{
  __shared__ float red[8];
  const size_t base = ((size_t)blockIdx.x * 256 + threadIdx.x) * 8;
  const int b   = (int)(base >> 20);
  const int rem = (int)(base & 1048575u);
  const int i   = rem >> 10;
  const int m   = rem & 1023;
  const float ui = u[b * SEQ + i];
  const float* vb = v + b * MVID;
  float acc = 0.0f;
  #pragma unroll
  for (int j = 0; j < 8; ++j) {
    const float kv = Kmat[base + j];
    const float tv = ui * kv * vb[m + j];
    out[1 + base + j] = tv;
    acc += tv * (-REGc * __logf(kv));    // C = -REG*log(K) exactly inverts K=exp(-C/REG)
  }
  #pragma unroll
  for (int off = 16; off > 0; off >>= 1) acc += __shfl_xor(acc, off, 32);
  if ((threadIdx.x & 31) == 0) red[threadIdx.x >> 5] = acc;
  __syncthreads();
  if (threadIdx.x == 0) {
    float s = 0.0f;
    #pragma unroll
    for (int w = 0; w < 8; ++w) s += red[w];
    atomicAdd(out, s * (1.0f / (float)BATCH));
  }
}

// ---------------------------------------------------------------------------
extern "C" void kernel_launch(void* const* d_in, const int* in_sizes, int n_in,
                              void* d_out, int out_size, void* d_ws, size_t ws_size,
                              hipStream_t stream) {
  const float* text  = (const float*)d_in[0];   // (16,1024,2560)
  const float* video = (const float*)d_in[1];   // (16,1024,2560)
  const float* tpr   = (const float*)d_in[2];   // (16,1024)
  const float* vpr   = (const float*)d_in[3];   // (16,1024)
  const float* Wt    = (const float*)d_in[4];   // (2560,256)
  const float* bt    = (const float*)d_in[5];   // (256)
  const float* Wv    = (const float*)d_in[6];   // (2560,256)
  const float* bv    = (const float*)d_in[7];   // (256)
  float* out = (float*)d_out;                   // [loss, T(16,1024,1024)]

  char* ws = (char*)d_ws;
  const size_t PROJ_BYTES = (size_t)NROWS * ADIM * sizeof(float);      // 16.78 MB
  const size_t NORM_BYTES = (size_t)NROWS * ADIM * sizeof(_Float16);   //  8.39 MB
  const size_t KMAT_BYTES = (size_t)BATCH * SEQ * MVID * sizeof(float);// 67.1 MB
  const size_t UV_BYTES   = (size_t)2 * NROWS * sizeof(float);
  const size_t WT_BYTES   = (size_t)ADIM * DIM * sizeof(_Float16);     //  1.31 MB
  float*    Eproj = (float*)   (ws);
  float*    Vproj = (float*)   (ws + PROJ_BYTES);
  _Float16* En    = (_Float16*)(ws + 2*PROJ_BYTES);
  _Float16* Vn    = (_Float16*)(ws + 2*PROJ_BYTES + NORM_BYTES);
  float*    Kmat  = (float*)   (ws + 2*PROJ_BYTES + 2*NORM_BYTES);
  float*    uvec  = (float*)   (ws + 2*PROJ_BYTES + 2*NORM_BYTES + KMAT_BYTES);
  float*    vvec  = uvec + NROWS;
  _Float16* WTt   = (_Float16*)(ws + 2*PROJ_BYTES + 2*NORM_BYTES + KMAT_BYTES + UV_BYTES);
  _Float16* WTv   = (_Float16*)((char*)WTt + WT_BYTES);

  wtrans_kernel<<<(ADIM*DIM)/256, 256, 0, stream>>>(Wt, WTt);
  wtrans_kernel<<<(ADIM*DIM)/256, 256, 0, stream>>>(Wv, WTv);
  proj_kernel<<<1024, 256, 0, stream>>>(text,  tpr, WTt, bt, Eproj);
  proj_kernel<<<1024, 256, 0, stream>>>(video, vpr, WTv, bv, Vproj);
  norm_kernel<<<2048, 256, 0, stream>>>(Eproj, En);
  norm_kernel<<<2048, 256, 0, stream>>>(Vproj, Vn);
  sim_kernel<<<4096, 256, 0, stream>>>(En, Vn, Kmat);
  init_uv_kernel<<<64, 256, 0, stream>>>(uvec, vvec);
  for (int it = 0; it < 10; ++it) {
    sk_row_kernel<<<2048, 256, 0, stream>>>(Kmat, vvec, uvec);
    sk_col_kernel<<<256,  256, 0, stream>>>(Kmat, uvec, vvec);
  }
  zero_loss_kernel<<<1, 1, 0, stream>>>(out);
  finalize_kernel<<<8192, 256, 0, stream>>>(Kmat, uvec, vvec, out);
}
